// MultiHeadAttention_14894946582935
// MI455X (gfx1250) — matrix operand order repaired and checked by reading, not compiled, not run on hardware
//
#include <hip/hip_runtime.h>
#include <hip/hip_bf16.h>

// MHA forward for B=4, S=2048, D=1024, H=16, HD=64 on gfx1250 (MI455X).
// bf16 WMMA (v_wmma_f32_16x16x32_bf16) + fp32 accumulation, fp32 softmax.
// Attention K/V staging uses the Tensor Data Mover (tensor_load_to_lds) with
// descriptor-side LDS padding; completion via TENSORcnt.
// Workspace (bf16 elems): xbf[8M] | Wq..Wo[1M ea] | Q,K,V[8M ea] | attn[8M] = 88MB.

typedef __bf16 bf16_t;
typedef __attribute__((ext_vector_type(16))) __bf16 v16bf;
typedef __attribute__((ext_vector_type(8)))  float   v8f;
typedef __attribute__((ext_vector_type(4)))  unsigned int u32x4;
typedef __attribute__((ext_vector_type(8)))  int          i32x8;
typedef __attribute__((ext_vector_type(4)))  int          i32x4;

#define WMMA_BF16(A, B, C) \
  __builtin_amdgcn_wmma_f32_16x16x32_bf16(false, (A), false, (B), (short)0, (C), false, false)

__device__ __forceinline__ v16bf ld16(const bf16_t* p) {
  v16bf r;
#pragma unroll
  for (int i = 0; i < 16; ++i) r[i] = p[i];
  return r;
}

// A-fragment helper: elements 0..7 from p0, 8..15 from p1 (K and K+16 halves).
__device__ __forceinline__ v16bf ld8x2(const bf16_t* p0, const bf16_t* p1) {
  v16bf r;
#pragma unroll
  for (int i = 0; i < 8; ++i) { r[i] = p0[i]; r[i + 8] = p1[i]; }
  return r;
}

// TDM: load a 2D tile (32 rows x 64 bf16, row stride 64 elems in memory) into
// LDS at lds_off, padding +8 DWORDs after every 32 DWORDs (row) -> LDS row
// stride 80 bf16. D# per CDNA5 ISA ch.8: group0 = count/lds/global/type,
// group1 = data_size|pad | dims/strides. Groups 2/3 zero (2D tensor).
__device__ __forceinline__ void tdm_load_tile_32x64(unsigned lds_off,
                                                    const bf16_t* gsrc) {
  unsigned long long ga = (unsigned long long)(const void*)gsrc;
  u32x4 g0 = { 1u,                                   // count=1, no gather
               lds_off,                               // lds_addr (bytes)
               (unsigned)(ga & 0xFFFFFFFFu),
               (unsigned)((ga >> 32) & 0x1FFFFFFu) | (2u << 30) };  // type=2
  i32x8 g1 = { (1 << 16) | (1 << 20) | (4 << 22) | (7 << 25),
               //  data_size=2B | pad_en | pad_interval=32dw | pad_amount=8dw
               (64 << 16),        // atomic_addr=0 | tensor_dim0 = 64 (lo16)
               (2048 << 16),      // tensor_dim0 hi=0 | tensor_dim1 = 2048 (lo16)
               (64 << 16),        // tensor_dim1 hi=0 | tile_dim0 = 64
               32,                // tile_dim1 = 32 | tile_dim2 = 0
               64,                // tensor_dim0_stride lo32 = 64 elems
               0, 0 };            // stride hi / dim1_stride
  i32x4 z4 = { 0, 0, 0, 0 };
#if defined(__clang_major__) && (__clang_major__ >= 23)
  i32x8 z8 = { 0, 0, 0, 0, 0, 0, 0, 0 };
  __builtin_amdgcn_tensor_load_to_lds(g0, g1, z4, z4, z8, 0);
#else
  __builtin_amdgcn_tensor_load_to_lds(g0, g1, z4, z4, 0);
#endif
}

// ---------------------------------------------------------------------------
// fp32 -> bf16 cast
// ---------------------------------------------------------------------------
__global__ void mha_cast_bf16(const float* __restrict__ src,
                              bf16_t* __restrict__ dst, int n) {
  int i = blockIdx.x * blockDim.x + threadIdx.x;
  if (i < n) dst[i] = (bf16_t)src[i];
}

// ---------------------------------------------------------------------------
// Y = X @ W^T (+ bias). X:[M,1024] bf16, W:[1024,1024] bf16.
// mode 0: bf16 head-split [B,H,S,HD] store; mode 1: fp32 flat + bias.
// Block 256 thr (8 waves), macro-tile 128M x 128N; wave = 32x64 (2x4 WMMA).
// ---------------------------------------------------------------------------
__global__ __launch_bounds__(256)
void mha_gemm_bf16(const bf16_t* __restrict__ X, const bf16_t* __restrict__ W,
                   const float* __restrict__ bias, bf16_t* __restrict__ Ybf,
                   float* __restrict__ Yf, int mode) {
  const int D_ = 1024;
  __shared__ bf16_t Xs[128 * 40];   // 128 rows x 32 k, stride 40 (16B aligned)
  __shared__ bf16_t Ws[128 * 40];

  const int mB = blockIdx.x * 128;
  const int nB = blockIdx.y * 128;
  const int tid  = threadIdx.x;
  const int wave = tid >> 5, lane = tid & 31;
  const int wm = wave & 3, wn = wave >> 2;   // 4 waves in M, 2 in N
  const int nl = lane & 15, hi = lane >> 4;

  v8f zero = {};
  v8f acc[2][4];
#pragma unroll
  for (int a = 0; a < 2; ++a)
#pragma unroll
    for (int b = 0; b < 4; ++b) acc[a][b] = zero;

  for (int k0 = 0; k0 < D_; k0 += 32) {
    __syncthreads();
    {
      const int row = tid >> 1, part = tid & 1;           // 2 thr/row, 32B each
      const int4* g = (const int4*)(X + (size_t)(mB + row) * D_ + k0 + part * 16);
      int4* l = (int4*)&Xs[row * 40 + part * 16];
      l[0] = g[0]; l[1] = g[1];
      const int4* gw = (const int4*)(W + (size_t)(nB + row) * D_ + k0 + part * 16);
      int4* lw = (int4*)&Ws[row * 40 + part * 16];
      lw[0] = gw[0]; lw[1] = gw[1];
    }
    __syncthreads();

    const int e0 = hi * 8;
    v16bf aF[2], bF[4];
#pragma unroll
    for (int ms = 0; ms < 2; ++ms) {
      const bf16_t* p = &Xs[(wm * 32 + ms * 16 + nl) * 40 + e0];
      aF[ms] = ld8x2(p, p + 16);                 // K {0-7,16-23} / {8-15,24-31}
    }
#pragma unroll
    for (int ns = 0; ns < 4; ++ns) {
      const bf16_t* p = &Ws[(wn * 64 + ns * 16 + nl) * 40 + hi * 8];
      bF[ns] = ld8x2(p, p + 16);                 // K {0-7,16-23} / {8-15,24-31}
    }
#pragma unroll
    for (int ms = 0; ms < 2; ++ms)
#pragma unroll
      for (int ns = 0; ns < 4; ++ns)
        acc[ms][ns] = WMMA_BF16(aF[ms], bF[ns], acc[ms][ns]);
  }

#pragma unroll
  for (int ms = 0; ms < 2; ++ms)
#pragma unroll
    for (int ns = 0; ns < 4; ++ns)
#pragma unroll
      for (int r = 0; r < 8; ++r) {
        const int m = mB + wm * 32 + ms * 16 + r + hi * 8;
        const int n = nB + wn * 64 + ns * 16 + nl;
        const float v = acc[ms][ns][r];
        if (mode == 0) {
          const int b = m >> 11, s = m & 2047, h = n >> 6, hd = n & 63;
          Ybf[(((size_t)(b * 16 + h)) * 2048 + s) * 64 + hd] = (bf16_t)v;
        } else {
          Yf[(size_t)m * D_ + n] = v + bias[n];
        }
      }
}

// ---------------------------------------------------------------------------
// Flash attention over head-split bf16 Q/K/V [B,H,S,HD]. Output bf16 [B*S,D].
// Block: 256 thr (8 waves) = 128 queries of one (b,h); wave owns 16 queries.
// K/V chunks (32 keys) staged into padded LDS by the Tensor Data Mover.
// Scores computed transposed (S^T = K_tile x Q^T) so softmax key-reduction is
// in-lane + one shfl_xor(16); exp(P) lands directly in the A-frag layout for
// O += P^T x V.
// ---------------------------------------------------------------------------
__global__ __launch_bounds__(256)
void mha_flash_attn(const bf16_t* __restrict__ Q, const bf16_t* __restrict__ K,
                    const bf16_t* __restrict__ V, bf16_t* __restrict__ O) {
  const int S_ = 2048, HD_ = 64;
  __shared__ bf16_t Kt[32 * 80];   // 32 keys x 64 hd, stride 80 (TDM-padded)
  __shared__ bf16_t Vt[32 * 80];

  const int bh  = blockIdx.y;                  // 0..63 = b*16+h
  const int tq  = blockIdx.x;                  // 0..15
  const int tid = threadIdx.x;
  const int wave = tid >> 5, lane = tid & 31;
  const int nl = lane & 15, hi = lane >> 4;
  const int qb = tq * 128 + wave * 16;
  const int qg = qb + nl;                      // this lane's query (score col)
  const size_t base = (size_t)bh * S_ * HD_;

  // LDS byte offsets of the tiles (flat LDS address low 32 bits = LDS offset).
  const unsigned ldsK = (unsigned)(unsigned long long)(const void*)&Kt[0];
  const unsigned ldsV = (unsigned)(unsigned long long)(const void*)&Vt[0];

  // Q fragments (B operand of score WMMA): lane nl reads row qg.
  const bf16_t* qp = Q + base + (size_t)qg * HD_ + hi * 8;
  const v16bf qB0 = ld8x2(qp, qp + 16);        // hd 0..31 chunk
  const v16bf qB1 = ld8x2(qp + 32, qp + 48);   // hd 32..63 chunk

  v8f zero = {};
  v8f acc[4];
#pragma unroll
  for (int i = 0; i < 4; ++i) acc[i] = zero;
  float mrun = -1e30f, lrun = 0.f;

  const int nch = tq * 4 + 4;                  // causal: keys 0 .. tq*128+127
  for (int c = 0; c < nch; ++c) {
    const int kb = c * 32;
    __syncthreads();                           // previous chunk's compute done
    if (wave == 0) {                           // TDM staging, one wave issues
      tdm_load_tile_32x64(ldsK, K + base + (size_t)kb * HD_);
      tdm_load_tile_32x64(ldsV, V + base + (size_t)kb * HD_);
      __builtin_amdgcn_s_wait_tensorcnt((short)0);
    }
    __syncthreads();                           // data visible to all waves
    if (kb > qb + 15) continue;                // chunk fully masked for wave

    const int e0 = hi * 8;
    // Scores S^T for two 16-key subtiles: D[m=key][n=query], fp32.
    v8f st[2];
#pragma unroll
    for (int t = 0; t < 2; ++t) {
      const bf16_t* kp = &Kt[(t * 16 + nl) * 80 + e0];
      const v16bf a0 = ld8x2(kp, kp + 16);         // hd 0..31
      const v16bf a1 = ld8x2(kp + 32, kp + 48);    // hd 32..63
      v8f s = zero;
      s = WMMA_BF16(a0, qB0, s);
      s = WMMA_BF16(a1, qB1, s);
      st[t] = s;
    }

    // Scale 1/sqrt(64), causal mask, online softmax (keys split across pair).
    float mx = mrun;
#pragma unroll
    for (int t = 0; t < 2; ++t)
#pragma unroll
      for (int r = 0; r < 8; ++r) {
        const int kg = kb + t * 16 + r + e0;       // key index of element
        float sv = st[t][r] * 0.125f;
        sv = (kg <= qg) ? sv : -1e30f;
        st[t][r] = sv;
        mx = fmaxf(mx, sv);
      }
    mx = fmaxf(mx, __shfl_xor(mx, 16, 32));
    const float alpha = __expf(mrun - mx);
    mrun = mx;
    float ls = 0.f;
#pragma unroll
    for (int t = 0; t < 2; ++t)
#pragma unroll
      for (int r = 0; r < 8; ++r) {
        const float p = __expf(st[t][r] - mx);
        st[t][r] = p;
        ls += p;
      }
    lrun = lrun * alpha + ls;

    // Rescale O accumulators: row m = r + hi*8 needs alpha of query m.
#pragma unroll
    for (int r = 0; r < 8; ++r) {
      const float am = __shfl(alpha, r + hi * 8, 32);
#pragma unroll
      for (int ns = 0; ns < 4; ++ns) acc[ns][r] *= am;
    }

    // Pack exp(P) into A-fragment: elems 0..7 = keys kb+{0..7|8..15},
    // elems 8..15 = keys kb+16+{0..7|8..15} — matches st[0]/st[1] directly.
    v16bf pA;
#pragma unroll
    for (int r = 0; r < 8; ++r) {
      pA[r]     = (bf16_t)st[0][r];
      pA[r + 8] = (bf16_t)st[1][r];
    }

    // O[q][d] += P^T x V : B operand = V columns (scalar LDS gather).
    const int koff = hi * 8;
#pragma unroll
    for (int ns = 0; ns < 4; ++ns) {
      v16bf vB;
#pragma unroll
      for (int e = 0; e < 16; ++e)
        vB[e] = Vt[(koff + e + (e & 8)) * 80 + ns * 16 + nl];
      acc[ns] = WMMA_BF16(pA, vB, acc[ns]);
    }
  }

  // Epilogue: combine l across lane pair, normalize, store merged-head bf16.
  const float lt  = lrun + __shfl_xor(lrun, 16, 32);
  const float inv = 1.f / lt;
  const int b = bh >> 4, h = bh & 15;
#pragma unroll
  for (int r = 0; r < 8; ++r) {
    const float iv = __shfl(inv, r + hi * 8, 32);
    const int m = qb + r + hi * 8;
    const size_t orow = ((size_t)(b * S_ + m)) * 1024 + h * 64;
#pragma unroll
    for (int ns = 0; ns < 4; ++ns)
      O[orow + ns * 16 + nl] = (bf16_t)(acc[ns][r] * iv);
  }
}

// ---------------------------------------------------------------------------
extern "C" void kernel_launch(void* const* d_in, const int* in_sizes, int n_in,
                              void* d_out, int out_size, void* d_ws, size_t ws_size,
                              hipStream_t stream) {
  (void)in_sizes; (void)n_in; (void)out_size; (void)ws_size;
  const float* x  = (const float*)d_in[0];
  const float* Wq = (const float*)d_in[1];
  const float* Wk = (const float*)d_in[2];
  const float* Wv = (const float*)d_in[3];
  const float* Wo = (const float*)d_in[4];
  const float* bo = (const float*)d_in[5];
  float* out = (float*)d_out;

  const size_t M = 4 * 2048, D = 1024;   // 8192 rows
  bf16_t* ws = (bf16_t*)d_ws;
  size_t o = 0;
  bf16_t* xbf = ws + o; o += M * D;
  bf16_t* wqb = ws + o; o += D * D;
  bf16_t* wkb = ws + o; o += D * D;
  bf16_t* wvb = ws + o; o += D * D;
  bf16_t* wob = ws + o; o += D * D;
  bf16_t* qb  = ws + o; o += M * D;      // [B,H,S,HD]
  bf16_t* kb  = ws + o; o += M * D;
  bf16_t* vb  = ws + o; o += M * D;
  bf16_t* ab  = ws + o; o += M * D;      // attn out, merged heads [B*S, D]

  const int nx = (int)(M * D), nw = (int)(D * D);
  mha_cast_bf16<<<(nx + 255) / 256, 256, 0, stream>>>(x,  xbf, nx);
  mha_cast_bf16<<<(nw + 255) / 256, 256, 0, stream>>>(Wq, wqb, nw);
  mha_cast_bf16<<<(nw + 255) / 256, 256, 0, stream>>>(Wk, wkb, nw);
  mha_cast_bf16<<<(nw + 255) / 256, 256, 0, stream>>>(Wv, wvb, nw);
  mha_cast_bf16<<<(nw + 255) / 256, 256, 0, stream>>>(Wo, wob, nw);

  dim3 gg(64, 8);                        // M/128 x N/128
  mha_gemm_bf16<<<gg, 256, 0, stream>>>(xbf, wqb, nullptr, qb, nullptr, 0);
  mha_gemm_bf16<<<gg, 256, 0, stream>>>(xbf, wkb, nullptr, kb, nullptr, 0);
  mha_gemm_bf16<<<gg, 256, 0, stream>>>(xbf, wvb, nullptr, vb, nullptr, 0);

  dim3 ga(16, 64);                       // S/128 x B*H
  mha_flash_attn<<<ga, 256, 0, stream>>>(qb, kb, vb, ab);

  mha_gemm_bf16<<<gg, 256, 0, stream>>>(ab, wob, bo, nullptr, out, 1);
}
